// QConv2d_82471962017915
// MI455X (gfx1250) — 8-bit matrix operand order repaired and checked by reading, compiled, not run on hardware
//
#include <hip/hip_runtime.h>

typedef __attribute__((ext_vector_type(8))) int v8i;

#define NIMG  32
#define CIN   128
#define HW    56
#define COUT  256
#define PIX   (HW*HW)
#define MTOT  (NIMG*PIX)
#define XQ_BYTES ((size_t)NIMG*HW*HW*CIN)

#define LDS_ROW_STRIDE 144
#define LDS_BUF_BYTES  (64 * LDS_ROW_STRIDE)

__global__ __launch_bounds__(256) void quantize_x_kernel(
    const float* __restrict__ x, signed char* __restrict__ xq)
{
    __shared__ __attribute__((aligned(4))) signed char tile[HW * CIN];
    const int nh = blockIdx.x;
    const int n  = nh / HW;
    const int h  = nh % HW;
    const float* src = x + (size_t)n * CIN * PIX + (size_t)h * HW;
    const int t = threadIdx.x;

    for (int idx = t; idx < HW * CIN; idx += 256) {
        int c = idx / HW;
        int w = idx % HW;
        float f = src[(size_t)c * PIX + w];
        tile[w * CIN + c] = (signed char)(int)f;
    }
    __syncthreads();

    unsigned int* dst = (unsigned int*)(xq + (size_t)nh * HW * CIN);
    const unsigned int* tl = (const unsigned int*)tile;
    for (int idx = t; idx < (HW * CIN) / 4; idx += 256)
        dst[idx] = tl[idx];
}

__global__ __launch_bounds__(256) void quantize_w_kernel(
    const float* __restrict__ w, signed char* __restrict__ wq)
{
    int id = blockIdx.x * 256 + threadIdx.x;
    if (id >= 9 * COUT * CIN) return;
    int c  = id & 127;
    int o  = (id >> 7) & 255;
    int rs = id >> 15;
    float f = w[((size_t)o * CIN + c) * 9 + rs];
    wq[id] = (signed char)(int)f;
}

__global__ __launch_bounds__(256) void qconv_wmma_kernel(
    const signed char* __restrict__ xq,
    const signed char* __restrict__ wq,
    const float* __restrict__ bias,
    const int*   __restrict__ Aq,
    const int*   __restrict__ Nq,
    const int*   __restrict__ pmin,
    const int*   __restrict__ pmax,
    float*       __restrict__ out)
{
    __shared__ __attribute__((aligned(16))) signed char lds_a[2][LDS_BUF_BYTES];

    const int t      = threadIdx.x;
    const int lane   = t & 31;
    const int wave   = t >> 5;
    const int m_sub  = wave & 3;
    const int n_half = wave >> 2;
    const int n_lo   = lane & 15;
    const int khalf  = lane >> 4;

    const int m_base = blockIdx.x * 64;
    const int n_base = blockIdx.y * 128 + n_half * 64;

    const int row = t >> 2;
    const int seg = t & 3;
    const int m_ld   = m_base + row;
    const int ni_ld  = m_ld / PIX;
    const int rem_ld = m_ld % PIX;
    const int p_ld   = rem_ld / HW;
    const int q_ld   = rem_ld % HW;

    auto issue_stage = [&](int rs, int buf) {
        const int r  = rs / 3, s = rs % 3;
        const int hh = p_ld + r - 1;
        const int ww = q_ld + s - 1;
        signed char* lp = &lds_a[buf][0] + row * LDS_ROW_STRIDE + seg * 32;
        if (hh >= 0 && hh < HW && ww >= 0 && ww < HW) {
            const signed char* g =
                xq + ((size_t)((ni_ld * HW + hh) * HW + ww)) * CIN + seg * 32;
            unsigned l32 = (unsigned)(unsigned long long)lp;
            asm volatile("global_load_async_to_lds_b128 %0, %1, off"
                         :: "v"(l32), "v"(g) : "memory");
            asm volatile("global_load_async_to_lds_b128 %0, %1, off"
                         :: "v"(l32 + 16u), "v"(g + 16) : "memory");
        } else {
            int4 z = make_int4(0, 0, 0, 0);
            *(int4*)(lp)      = z;
            *(int4*)(lp + 16) = z;
        }
    };

    v8i acc[4];
#pragma unroll
    for (int i = 0; i < 4; ++i) {
        v8i z = {0,0,0,0,0,0,0,0};
        acc[i] = z;
    }

    issue_stage(0, 0);

    for (int rs = 0; rs < 9; ++rs) {
        const int cur = rs & 1;
        if (rs < 8) {
            issue_stage(rs + 1, cur ^ 1);
            asm volatile("s_wait_asynccnt 0x2" ::: "memory");
        } else {
            asm volatile("s_wait_asynccnt 0x0" ::: "memory");
        }
        __syncthreads();

        const signed char* abase =
            &lds_a[cur][0] + (m_sub * 16 + n_lo) * LDS_ROW_STRIDE;

#pragma unroll
        for (int ck = 0; ck < 2; ++ck) {
            const signed char* ap = abase + ck * 64 + khalf * 16;
            int2 a0 = *(const int2*)(ap);
            int2 a1 = *(const int2*)(ap + 8);
            int2 a2 = *(const int2*)(ap + 32);
            int2 a3 = *(const int2*)(ap + 40);
            v8i av;
            av[0] = a0.x; av[1] = a0.y; av[2] = a1.x; av[3] = a1.y;
            av[4] = a2.x; av[5] = a2.y; av[6] = a3.x; av[7] = a3.y;

#pragma unroll
            for (int nt = 0; nt < 4; ++nt) {
                const signed char* bp =
                    wq + ((size_t)(rs * COUT + n_base + nt * 16 + n_lo)) * CIN
                       + ck * 64 + khalf * 16;
                int4 bl = *(const int4*)(bp);
                int4 bh = *(const int4*)(bp + 32);
                v8i bv;
                bv[0] = bl.x; bv[1] = bl.y; bv[2] = bl.z; bv[3] = bl.w;
                bv[4] = bh.x; bv[5] = bh.y; bv[6] = bh.z; bv[7] = bh.w;

                acc[nt] = __builtin_amdgcn_wmma_i32_16x16x64_iu8(
                     true, av,  true, bv,
                    acc[nt],  false,  false);
            }
        }
        __syncthreads();
    }

    const float lo = (float)pmin[0];
    const float hi = (float)pmax[0];

    int moff[8];
#pragma unroll
    for (int v = 0; v < 8; ++v) {
        int m  = m_base + m_sub * 16 + khalf * 8 + v;
        int ni = m / PIX;
        int rm = m % PIX;
        moff[v] = ni * (COUT * PIX) + rm;
    }

#pragma unroll
    for (int nt = 0; nt < 4; ++nt) {
        const int o = n_base + nt * 16 + n_lo;
        const float bf = bias[o];
        const float Af = (float)Aq[o];
        const int   Nv = Nq[o];
#pragma unroll
        for (int v = 0; v < 8; ++v) {
            float rr = rintf((float)acc[nt][v] + bf) * Af;
            rr = ldexpf(rr, -Nv);
            rr = rintf(rr);
            rr = fminf(fmaxf(rr, lo), hi);
            out[(size_t)moff[v] + (size_t)o * PIX] = rr;
        }
    }
}

extern "C" void kernel_launch(void* const* d_in, const int* in_sizes, int n_in,
                              void* d_out, int out_size, void* d_ws, size_t ws_size,
                              hipStream_t stream)
{
    const float* x    = (const float*)d_in[0];
    const float* w    = (const float*)d_in[1];
    const float* bias = (const float*)d_in[2];
    const int*   Aq   = (const int*)d_in[3];
    const int*   Nq   = (const int*)d_in[4];
    const int*   pmin = (const int*)d_in[5];
    const int*   pmax = (const int*)d_in[6];

    signed char* xq = (signed char*)d_ws;
    signed char* wp = xq + XQ_BYTES;

    quantize_x_kernel<<<NIMG * HW, 256, 0, stream>>>(x, xq);
    quantize_w_kernel<<<(9 * COUT * CIN + 255) / 256, 256, 0, stream>>>(w, wp);

    dim3 grid(MTOT / 64, 2);
    qconv_wmma_kernel<<<grid, 256, 0, stream>>>(
        xq, wp, bias, Aq, Nq, pmin, pmax, (float*)d_out);
}
